// SingleSolveCDE_51951924412753
// MI455X (gfx1250) — matrix operand order repaired and checked by reading, not compiled, not run on hardware
//
#include <hip/hip_runtime.h>
#include <hip/hip_bf16.h>
#include <math.h>

#define L 100
#define BATCH 512
#define WIDTH 128
#define HID 3
#define NSTEPS ((L - 1) * 5)   /* 495 */
#define BT 16                  /* batch rows per workgroup */
#define NWG (BATCH / BT)       /* 32 workgroups */

typedef __attribute__((ext_vector_type(16))) __bf16 v16bf;
typedef __attribute__((ext_vector_type(8)))  float  v8f;
typedef __attribute__((ext_vector_type(16))) unsigned short v16u;

// ---- Tsit5 coefficients -----------------------------------------------------
#define A21f 0.161f
#define A31f (-0.008480655492356989f)
#define A32f 0.335480655492357f
#define A41f 2.8971530571054935f
#define A42f (-6.359448489975075f)
#define A43f 4.3622954328695815f
#define A51f 5.325864828439257f
#define A52f (-11.748883564062828f)
#define A53f 7.4955393428898365f
#define A54f (-0.09249506636175525f)
#define A61f 5.86145544294642f
#define A62f (-12.92096931784711f)
#define A63f 8.159367898576159f
#define A64f (-0.071584973281401f)
#define A65f (-0.028269050394068383f)
#define B1f 0.09646076681806523f
#define B2f 0.01f
#define B3f 0.4798896504144996f
#define B4f 1.379008574103742f
#define B5f (-3.290069515436081f)
#define B6f 2.324710524099774f
#define C2f 0.161f
#define C3f 0.327f
#define C4f 0.9f
#define C5f 0.9800255409045097f

#define LOG2Ef 1.4426950408889634f
#define LN2f   0.6931471805599453f

__device__ __forceinline__ unsigned short f2bf(float f) {
  unsigned int u = __float_as_uint(f);
  unsigned int r = u + 0x7FFFu + ((u >> 16) & 1u);   // round-to-nearest-even
  return (unsigned short)(r >> 16);
}

// softplus via bare v_exp_f32/v_log_f32 (exp2 arg <= 0 -> result in (0,1];
// log2 arg in [1,2] -> no denorm/range-reduction corner cases).
__device__ __forceinline__ float softplus_f(float x) {
  float e = __builtin_amdgcn_exp2f(-fabsf(x) * LOG2Ef);
  return fmaxf(x, 0.0f) + LN2f * __builtin_amdgcn_logf(1.0f + e);
}

// tanh = 1 - 2/(exp(2x)+1) via v_exp_f32 + v_rcp_f32; saturates at +-1.
__device__ __forceinline__ float tanh_f(float x) {
  float e = __builtin_amdgcn_exp2f(x * (2.0f * LOG2Ef));
  return 1.0f - 2.0f * __builtin_amdgcn_rcpf(e + 1.0f);
}

// Load 16 bf16 (as ushort bits) from two 16-byte LDS chunks into one 32B vector.
__device__ __forceinline__ v16u ld2x16(const unsigned short* p0, const unsigned short* p1) {
  union { v16u v; uint4 q[2]; } u;
  u.q[0] = *(const uint4*)p0;
  u.q[1] = *(const uint4*)p1;
  return u.v;
}

// One MLP evaluation for the 16-row tile: sYe[16][3] -> sP[4][16][16] (pre-tanh
// layer-2 partials, one K-tile per partial; bias/tanh applied by combiner).
__device__ __forceinline__ void mlp_eval(
    const float* sYe, const float* sW0, const float* sB0, const float* sB1,
    const unsigned short* sW1T, const unsigned short* sW2T,
    unsigned short* sH0, unsigned short* sH1, float* sP,
    int tid, int lane, int wv)
{
  // ---- layer 0: (16x3) @ W0^T + b0, softplus, stored bf16 row-major [16][128]
  for (int i = tid; i < BT * WIDTH; i += 256) {
    int bl = i >> 7, j = i & 127;
    const float* y = &sYe[bl * HID];
    float a = sB0[j] + y[0] * sW0[j * HID + 0] + y[1] * sW0[j * HID + 1]
                     + y[2] * sW0[j * HID + 2];
    sH0[bl * WIDTH + j] = f2bf(softplus_f(a));
  }
  __syncthreads();

  // ---- layer 1: (16x128) @ W1^T + b1 via WMMA; wave wv owns N-tile wv
  {
    const int m  = lane & 15;     // A-fragment row
    const int hi = lane >> 4;     // A-fragment K-half selector
    const int n  = lane & 15;     // D-fragment column
    v8f acc;
    float bb = sB1[wv * 16 + n];
#pragma unroll
    for (int r = 0; r < 8; ++r) acc[r] = bb;
#pragma unroll
    for (int kt = 0; kt < 4; ++kt) {
      const int kbase = kt * 32 + hi * 8;
      v16u au = ld2x16(&sH0[m * WIDTH + kbase], &sH0[m * WIDTH + kbase + 16]);
      const unsigned short* bp = &sW1T[(wv * 16 + n) * WIDTH + kbase];
      v16u bu = ld2x16(bp, bp + 16);
      acc = __builtin_amdgcn_wmma_f32_16x16x32_bf16(
          false, __builtin_bit_cast(v16bf, au),
          false, __builtin_bit_cast(v16bf, bu),
          (short)0, acc, false, false);
    }
#pragma unroll
    for (int r = 0; r < 8; ++r) {
      int mm = r + 8 * hi;
      sH1[mm * WIDTH + wv * 16 + n] = f2bf(softplus_f(acc[r]));
    }
  }
  __syncthreads();

  // ---- layer 2: (16x128) @ W2^T (padded 9->16 cols); waves 0..3 do one K-tile
  // each into f32 partials; no dependent WMMA chain on the critical path.
  if (wv < 4) {
    const int m  = lane & 15;
    const int hi = lane >> 4;
    const int n  = lane & 15;
    const int kt = wv;
    v8f acc;
#pragma unroll
    for (int r = 0; r < 8; ++r) acc[r] = 0.0f;
    const int kbase = kt * 32 + hi * 8;
    v16u au = ld2x16(&sH1[m * WIDTH + kbase], &sH1[m * WIDTH + kbase + 16]);
    const unsigned short* bp = &sW2T[n * WIDTH + kbase];
    v16u bu = ld2x16(bp, bp + 16);
    acc = __builtin_amdgcn_wmma_f32_16x16x32_bf16(
        false, __builtin_bit_cast(v16bf, au),
        false, __builtin_bit_cast(v16bf, bu),
        (short)0, acc, false, false);
#pragma unroll
    for (int r = 0; r < 8; ++r) {
      int mm = r + 8 * hi;
      sP[kt * (BT * 16) + mm * 16 + n] = acc[r];
    }
  }
  __syncthreads();
}

// k[b][c] = sum_j tanh(b2 + sum_kt P[kt])[c*3+j] * dX[b][idx(t)][j]
__device__ __forceinline__ float fvec_k(
    const float* sP, const float* sB2, const float* dxp, int bl, int c, float tstage)
{
  int idx = (int)floorf(tstage * (float)(L - 1));
  idx = idx < 0 ? 0 : (idx > L - 2 ? L - 2 : idx);
  const float* dx = dxp + idx * HID;
  float kv = 0.0f;
#pragma unroll
  for (int j = 0; j < 3; ++j) {
    int n = c * 3 + j;
    float a = sB2[n] + sP[bl * 16 + n] + sP[256 + bl * 16 + n]
            + sP[512 + bl * 16 + n] + sP[768 + bl * 16 + n];
    kv += tanh_f(a) * dx[j];
  }
  return kv;
}

__global__ __launch_bounds__(256)
void cde_kernel(const float* __restrict__ ts, const float* __restrict__ ys,
                const float* __restrict__ saveat,
                const float* __restrict__ W0, const float* __restrict__ b0,
                const float* __restrict__ W1, const float* __restrict__ b1,
                const float* __restrict__ W2, const float* __restrict__ b2,
                const int* __restrict__ cu_p, const int* __restrict__ tu_p,
                float* __restrict__ out)
{
  __shared__ alignas(16) unsigned short sW1T[WIDTH * WIDTH]; // W1 transposed, bf16
  __shared__ alignas(16) unsigned short sW2T[WIDTH * 16];    // W2 transposed+padded
  __shared__ alignas(16) unsigned short sH0[BT * WIDTH];
  __shared__ alignas(16) unsigned short sH1[BT * WIDTH];
  __shared__ float sW0[WIDTH * HID];
  __shared__ float sB0[WIDTH];
  __shared__ float sB1[WIDTH];
  __shared__ float sB2[16];
  __shared__ float sdX[BT * (L - 1) * HID];
  __shared__ float sYe[BT * HID];
  __shared__ alignas(16) float sP[4 * BT * 16];              // layer-2 partials
  __shared__ alignas(16) float sTraj[(NSTEPS + 1) * BT];     // component 0 only

  const int tid  = threadIdx.x;
  const int wg   = blockIdx.x;
  const int lane = tid & 31;
  const int wv   = tid >> 5;
  const int cu   = *cu_p;
  const int tu   = *tu_p;
  const float t0 = ts[0];
  const float h  = (ts[1] - ts[0]) * 0.2f;

  // ---- setup: weights into LDS (W1/W2 pre-transposed to WMMA-B-friendly order)
  for (int i = tid; i < WIDTH * WIDTH; i += 256) {
    int n = i >> 7, k = i & 127;
    sW1T[n * WIDTH + k] = f2bf(W1[i]);
  }
  for (int i = tid; i < WIDTH * 16; i += 256) {
    int k = i >> 4, n = i & 15;
    sW2T[n * WIDTH + k] = (n < 9) ? f2bf(W2[n * WIDTH + k]) : (unsigned short)0;
  }
  for (int i = tid; i < WIDTH * HID; i += 256) sW0[i] = W0[i];
  for (int i = tid; i < WIDTH; i += 256) { sB0[i] = b0[i]; sB1[i] = b1[i]; }
  if (tid < 16) sB2[tid] = (tid < 9) ? b2[tid] : 0.0f;

  // ---- control path x[b][l] = ys[b][min(l,cu-1)] (identity interp on same grid)
  float* sX = sTraj;  // reuse trajectory buffer as scratch during setup
  for (int i = tid; i < BT * L; i += 256) {
    int bl = i / L, l = i % L;
    int ls = l < cu ? l : (cu - 1);
    sX[bl * L + l] = ys[(wg * BT + bl) * L + ls];
  }
  __syncthreads();

  // ---- dX[b][l][c] = (ys3[l+1]-ys3[l]) * (L-1)
  const float inv_seg = (float)(L - 1);
  for (int i = tid; i < BT * (L - 1); i += 256) {
    int bl = i / (L - 1), l = i % (L - 1);
    float tg0 = (float)l / (float)(L - 1);
    float tg1 = (float)(l + 1) / (float)(L - 1);
    float x0 = sX[bl * L + l], x1 = sX[bl * L + l + 1];
    float* d = &sdX[(bl * (L - 1) + l) * HID];
    d[0] = (x1 - x0) * inv_seg;
    d[1] = (tg1 - tg0) * inv_seg;
    d[2] = (x1 * tg1 - x0 * tg0) * inv_seg;
  }

  // ---- ODE state lives in registers of threads 0..47 (one (row,comp) each)
  const bool owner = (tid < BT * HID);
  const int  bl    = tid / HID;
  const int  c     = tid % HID;
  const float* dxp = &sdX[bl * (L - 1) * HID];
  float yreg = 0.0f;
  float k0 = 0.0f, k1 = 0.0f, k2 = 0.0f, k3 = 0.0f, k4 = 0.0f, k5 = 0.0f;
  if (owner) yreg = (c == 0) ? sX[bl * L + 0] : 0.0f;  // y0 = (x[0], 0, 0)
  __syncthreads();
  if (owner && c == 0) sTraj[bl] = yreg;  // traj[0] (overwrites dead sX)
  if (owner) sYe[tid] = yreg;             // stage-1 input for first step

  // ---- RK loop: 495 steps x 6 stages, all resident in LDS/registers
  for (int i = 0; i < NSTEPS; ++i) {
    float t = t0 + (float)i * h;

    __syncthreads();
    mlp_eval(sYe, sW0, sB0, sB1, sW1T, sW2T, sH0, sH1, sP, tid, lane, wv);
    if (owner) {
      k0 = fvec_k(sP, sB2, dxp, bl, c, t);
      sYe[tid] = yreg + h * (A21f * k0);
    }
    __syncthreads();
    mlp_eval(sYe, sW0, sB0, sB1, sW1T, sW2T, sH0, sH1, sP, tid, lane, wv);
    if (owner) {
      k1 = fvec_k(sP, sB2, dxp, bl, c, t + C2f * h);
      sYe[tid] = yreg + h * (A31f * k0 + A32f * k1);
    }
    __syncthreads();
    mlp_eval(sYe, sW0, sB0, sB1, sW1T, sW2T, sH0, sH1, sP, tid, lane, wv);
    if (owner) {
      k2 = fvec_k(sP, sB2, dxp, bl, c, t + C3f * h);
      sYe[tid] = yreg + h * (A41f * k0 + A42f * k1 + A43f * k2);
    }
    __syncthreads();
    mlp_eval(sYe, sW0, sB0, sB1, sW1T, sW2T, sH0, sH1, sP, tid, lane, wv);
    if (owner) {
      k3 = fvec_k(sP, sB2, dxp, bl, c, t + C4f * h);
      sYe[tid] = yreg + h * (A51f * k0 + A52f * k1 + A53f * k2 + A54f * k3);
    }
    __syncthreads();
    mlp_eval(sYe, sW0, sB0, sB1, sW1T, sW2T, sH0, sH1, sP, tid, lane, wv);
    if (owner) {
      k4 = fvec_k(sP, sB2, dxp, bl, c, t + C5f * h);
      sYe[tid] = yreg + h * (A61f * k0 + A62f * k1 + A63f * k2 + A64f * k3
                            + A65f * k4);
    }
    __syncthreads();
    mlp_eval(sYe, sW0, sB0, sB1, sW1T, sW2T, sH0, sH1, sP, tid, lane, wv);
    if (owner) {
      k5 = fvec_k(sP, sB2, dxp, bl, c, t + h);
      yreg += h * (B1f * k0 + B2f * k1 + B3f * k2 + B4f * k3 + B5f * k4
                 + B6f * k5);
      if (c == 0) sTraj[(i + 1) * BT + bl] = yreg;
      sYe[tid] = yreg;  // stage-1 input for next step
    }
  }
  __syncthreads();

  // ---- sample trajectory component 0 at saveat[0:tu], out[b][j] row-major
  for (int f = tid; f < BT * tu; f += 256) {
    int b2i = f / tu, j = f % tu;
    float pos = (saveat[j] - t0) / h;
    int idx = (int)floorf(pos);
    idx = idx < 0 ? 0 : (idx > NSTEPS - 1 ? NSTEPS - 1 : idx);
    float frac = pos - (float)idx;
    float v0 = sTraj[idx * BT + b2i];
    float v1 = sTraj[(idx + 1) * BT + b2i];
    out[(wg * BT + b2i) * tu + j] = v0 + frac * (v1 - v0);
  }
}

extern "C" void kernel_launch(void* const* d_in, const int* in_sizes, int n_in,
                              void* d_out, int out_size, void* d_ws, size_t ws_size,
                              hipStream_t stream) {
  const float* ts     = (const float*)d_in[0];
  const float* ys     = (const float*)d_in[1];
  const float* saveat = (const float*)d_in[2];
  const float* W0     = (const float*)d_in[3];
  const float* b0     = (const float*)d_in[4];
  const float* W1     = (const float*)d_in[5];
  const float* b1     = (const float*)d_in[6];
  const float* W2     = (const float*)d_in[7];
  const float* b2     = (const float*)d_in[8];
  const int*   cu     = (const int*)d_in[9];
  const int*   tu     = (const int*)d_in[10];
  (void)in_sizes; (void)n_in; (void)out_size; (void)d_ws; (void)ws_size;
  cde_kernel<<<NWG, 256, 0, stream>>>(ts, ys, saveat, W0, b0, W1, b1, W2, b2,
                                      cu, tu, (float*)d_out);
}
